// Head_5299989643574
// MI455X (gfx1250) — compile-verified, matrix operand order checked by reading, not run on hardware
//
#include <hip/hip_runtime.h>
#include <hip/hip_bf16.h>
#include <math.h>

typedef __attribute__((ext_vector_type(16))) __bf16 v16bf;
typedef __attribute__((ext_vector_type(8)))  float  v8f;

#define T_SEQ    256
#define C_DIM    384
#define H_DIM    64
#define NTHREADS 512
#define SMEM_BYTES 155648   // 32K Xs/P | 24K Wt | 32K Q | 32K K | 32K Vt

// Load one 16x32 bf16 WMMA operand fragment (A layout for row-major source,
// B layout when the source is stored "transposed" so each lane reads a row).
// Per lane: two contiguous 16B LDS reads covering k = g*16 + half*8 + [0..7].
__device__ __forceinline__ v16bf ld_frag(const __bf16* base, int row, int stride,
                                         int kbase, int lane) {
  const __bf16* p = base + row * stride + kbase + (((lane >> 4) & 1) << 3);
  union { v16bf v; uint4 q[2]; } u;
  u.q[0] = *(const uint4*)(p);
  u.q[1] = *(const uint4*)(p + 16);
  return u.v;
}

__global__ __launch_bounds__(NTHREADS)
void rope_attn_kernel(const float* __restrict__ x,  const float* __restrict__ cosT,
                      const float* __restrict__ sinT, const float* __restrict__ Wq,
                      const float* __restrict__ Wk,  const float* __restrict__ Wv,
                      float* __restrict__ out) {
  extern __shared__ __align__(16) char smem[];
  __bf16* Xs = (__bf16*)(smem);            // [256][64] bf16 chunk of x
  __bf16* Wt = (__bf16*)(smem + 32768);    // 3 x [64][64] bf16, transposed [h][k]
  __bf16* Qs = (__bf16*)(smem + 57344);    // [256][64] rope'd + pre-scaled
  __bf16* Ks = (__bf16*)(smem + 90112);    // [256][64] rope'd
  __bf16* Vt = (__bf16*)(smem + 122880);   // [64][256] transposed
  __bf16* Pb = (__bf16*)(smem);            // aliases Xs: per-wave [16][32]

  const int tid  = threadIdx.x;
  const int lane = tid & 31;
  const int wave = tid >> 5;
  const int n    = lane & 15;          // column (N) / row-within-tile (A's M)
  const int hw   = (lane >> 4) & 1;    // which half-wave
  const int b    = blockIdx.x;
  const int r0   = wave * 16;          // this wave's 16-row strip

  const float* Wmat[3] = {Wq, Wk, Wv};

  // ---------------- Phase 1: Q/K/V projections, K-chunked over C ----------
  v8f acc[3][4];
  #pragma unroll
  for (int w = 0; w < 3; ++w)
    #pragma unroll
    for (int nt = 0; nt < 4; ++nt)
      #pragma unroll
      for (int v = 0; v < 8; ++v) acc[w][nt][v] = 0.0f;

  for (int chunk = 0; chunk < C_DIM / 64; ++chunk) {
    const int c0 = chunk * 64;
    // stage x[b, :, c0:c0+64] -> bf16 LDS (float4 loads)
    for (int idx = tid; idx < 256 * 16; idx += NTHREADS) {
      const int row = idx >> 4, c4 = (idx & 15) << 2;
      float4 f = *(const float4*)(x + ((size_t)b * T_SEQ + row) * C_DIM + c0 + c4);
      Xs[row * 64 + c4 + 0] = (__bf16)f.x;
      Xs[row * 64 + c4 + 1] = (__bf16)f.y;
      Xs[row * 64 + c4 + 2] = (__bf16)f.z;
      Xs[row * 64 + c4 + 3] = (__bf16)f.w;
    }
    // stage W[c0:c0+64, :] transposed -> Wt[h][k]
    #pragma unroll
    for (int w = 0; w < 3; ++w) {
      const float* W = Wmat[w];
      __bf16* dst = Wt + w * 64 * 64;
      for (int idx = tid; idx < 64 * 16; idx += NTHREADS) {
        const int k = idx >> 4, h4 = (idx & 15) << 2;
        float4 f = *(const float4*)(W + (size_t)(c0 + k) * H_DIM + h4);
        dst[(h4 + 0) * 64 + k] = (__bf16)f.x;
        dst[(h4 + 1) * 64 + k] = (__bf16)f.y;
        dst[(h4 + 2) * 64 + k] = (__bf16)f.z;
        dst[(h4 + 3) * 64 + k] = (__bf16)f.w;
      }
    }
    __syncthreads();

    #pragma unroll
    for (int kk = 0; kk < 64; kk += 32) {
      v16bf af = ld_frag(Xs, r0 + n, 64, kk, lane);
      #pragma unroll
      for (int w = 0; w < 3; ++w)
        #pragma unroll
        for (int nt = 0; nt < 4; ++nt) {
          v16bf bf = ld_frag(Wt + w * 64 * 64, nt * 16 + n, 64, kk, lane);
          acc[w][nt] = __builtin_amdgcn_wmma_f32_16x16x32_bf16(
              false, af, false, bf, (short)0, acc[w][nt], false, false);
        }
    }
    __syncthreads();   // before next chunk overwrites Xs/Wt
  }

  // ---------------- RoPE in registers, stash Q/K/Vt in LDS ----------------
  // C-layout: lane holds column h = nt*16 + n; VGPR v holds row t = r0+v+8*hw.
  // rotate partner h +/- 32 lives in tile nt^2, same lane/vgpr.
  #pragma unroll
  for (int v = 0; v < 8; ++v) {
    const int t = r0 + v + 8 * hw;
    #pragma unroll
    for (int nt = 0; nt < 4; ++nt) {
      const int h = nt * 16 + n;
      const float cv = cosT[t * H_DIM + h];
      const float sv = sinT[t * H_DIM + h];
      const float q  = acc[0][nt][v];
      const float k  = acc[1][nt][v];
      const float qr = (nt < 2) ? -acc[0][nt + 2][v] : acc[0][nt - 2][v];
      const float kr = (nt < 2) ? -acc[1][nt + 2][v] : acc[1][nt - 2][v];
      Qs[t * 64 + h]   = (__bf16)(0.125f * (q * cv + qr * sv)); // fold 1/sqrt(H)
      Ks[t * 64 + h]   = (__bf16)(k * cv + kr * sv);
      Vt[h * 256 + t]  = (__bf16)(acc[2][nt][v]);
    }
  }
  __syncthreads();

  // ---------------- Phase 2: causal flash attention -----------------------
  __bf16* Pw = Pb + wave * (16 * 32);   // this wave's private P bounce buffer
  v16bf qf[2];
  qf[0] = ld_frag(Qs, r0 + n, 64, 0, lane);
  qf[1] = ld_frag(Qs, r0 + n, 64, 32, lane);

  v8f oacc[4];
  #pragma unroll
  for (int nt = 0; nt < 4; ++nt)
    #pragma unroll
    for (int v = 0; v < 8; ++v) oacc[nt][v] = 0.0f;
  float rowmax[8], rowsum[8];
  #pragma unroll
  for (int v = 0; v < 8; ++v) { rowmax[v] = -__builtin_inff(); rowsum[v] = 0.0f; }

  const int jmax = (r0 + 15) >> 5;
  for (int j = 0; j <= jmax; ++j) {
    const int s0 = j * 32;
    v8f sacc[2];
    #pragma unroll
    for (int tt = 0; tt < 2; ++tt)
      #pragma unroll
      for (int v = 0; v < 8; ++v) sacc[tt][v] = 0.0f;

    #pragma unroll
    for (int kk = 0; kk < 2; ++kk)
      #pragma unroll
      for (int tt = 0; tt < 2; ++tt) {
        v16bf kf = ld_frag(Ks, s0 + tt * 16 + n, 64, kk * 32, lane);
        sacc[tt] = __builtin_amdgcn_wmma_f32_16x16x32_bf16(
            false, qf[kk], false, kf, (short)0, sacc[tt], false, false);
      }

    // mask + online softmax; row stats reduced across the 16 lanes of a half
    #pragma unroll
    for (int v = 0; v < 8; ++v) {
      const int t = r0 + v + 8 * hw;
      float a0 = (s0 + n      <= t) ? sacc[0][v] : -__builtin_inff();
      float a1 = (s0 + 16 + n <= t) ? sacc[1][v] : -__builtin_inff();
      float mx = fmaxf(a0, a1);
      mx = fmaxf(mx, __shfl_xor(mx, 1, 32));
      mx = fmaxf(mx, __shfl_xor(mx, 2, 32));
      mx = fmaxf(mx, __shfl_xor(mx, 4, 32));
      mx = fmaxf(mx, __shfl_xor(mx, 8, 32));
      const float nm    = fmaxf(rowmax[v], mx);
      const float alpha = __expf(rowmax[v] - nm);
      rowmax[v] = nm;
      const float p0 = __expf(a0 - nm);
      const float p1 = __expf(a1 - nm);
      float ps = p0 + p1;
      ps += __shfl_xor(ps, 1, 32);
      ps += __shfl_xor(ps, 2, 32);
      ps += __shfl_xor(ps, 4, 32);
      ps += __shfl_xor(ps, 8, 32);
      rowsum[v] = rowsum[v] * alpha + ps;
      #pragma unroll
      for (int nt = 0; nt < 4; ++nt) oacc[nt][v] *= alpha;
      const int m = v + 8 * hw;
      Pw[m * 32 + n]      = (__bf16)p0;
      Pw[m * 32 + 16 + n] = (__bf16)p1;
    }
    // re-load P as an A fragment (wave-private; compiler inserts dscnt waits)
    v16bf pf = ld_frag(Pw, n, 32, 0, lane);
    #pragma unroll
    for (int nt = 0; nt < 4; ++nt) {
      v16bf vf = ld_frag(Vt, nt * 16 + n, 256, s0, lane);
      oacc[nt] = __builtin_amdgcn_wmma_f32_16x16x32_bf16(
          false, pf, false, vf, (short)0, oacc[nt], false, false);
    }
  }

  // ---------------- epilogue: normalize and store (coalesced) -------------
  #pragma unroll
  for (int v = 0; v < 8; ++v) {
    const float inv = 1.0f / rowsum[v];
    const int t = r0 + v + 8 * hw;
    #pragma unroll
    for (int nt = 0; nt < 4; ++nt)
      out[((size_t)b * T_SEQ + t) * H_DIM + nt * 16 + n] = oacc[nt][v] * inv;
  }
}

extern "C" void kernel_launch(void* const* d_in, const int* in_sizes, int n_in,
                              void* d_out, int out_size, void* d_ws, size_t ws_size,
                              hipStream_t stream) {
  const float* x    = (const float*)d_in[0];
  const float* cosT = (const float*)d_in[1];
  const float* sinT = (const float*)d_in[2];
  const float* Wq   = (const float*)d_in[3];
  const float* Wk   = (const float*)d_in[4];
  const float* Wv   = (const float*)d_in[5];
  float* out = (float*)d_out;

  const int Bn = in_sizes[0] / (T_SEQ * C_DIM);   // 1024
  (void)d_ws; (void)ws_size; (void)n_in; (void)out_size;

  hipFuncSetAttribute((const void*)rope_attn_kernel,
                      hipFuncAttributeMaxDynamicSharedMemorySize, SMEM_BYTES);
  rope_attn_kernel<<<Bn, NTHREADS, SMEM_BYTES, stream>>>(x, cosT, sinT, Wq, Wk, Wv, out);
}
